// AdaptiveScaledDotProductAttention_36524401885990
// MI455X (gfx1250) — matrix operand order repaired and checked by reading, not compiled, not run on hardware
//
#include <hip/hip_runtime.h>
#include <hip/hip_bf16.h>
#include <math.h>

typedef __bf16 bf16_t;
typedef __bf16 v16bf __attribute__((ext_vector_type(16)));
typedef __bf16 v8bf  __attribute__((ext_vector_type(8)));
typedef float  v8f   __attribute__((ext_vector_type(8)));
typedef float  f32x8 __attribute__((ext_vector_type(8)));

#define DM   1024
#define NH   16
#define DK   64
#define NROW 4096           // b*n = 4*1024
#define SCALE 0.125f        // 1/sqrt(64)

__device__ __forceinline__ v8f wmma_bf16(v16bf a, v16bf b, v8f c) {
  // D = A(16x32 bf16) * B(32x16 bf16) + C(16x16 f32)
  return __builtin_amdgcn_wmma_f32_16x16x32_bf16(
      /*neg_a=*/false, a, /*neg_b=*/false, b,
      /*c_mod=*/(short)0, c, /*reuse_a=*/false, /*reuse_b=*/false);
}

__device__ __forceinline__ v16bf concat8(v8bf lo, v8bf hi) {
  v16bf r;
#pragma unroll
  for (int i = 0; i < 8; ++i) { r[i] = lo[i]; r[i + 8] = hi[i]; }
  return r;
}

// ---------------------------------------------------------------------------
// Weight prep: W[k][n] f32 -> Wt[n][k] bf16, LDS-tiled so both global reads
// (along n) and global writes (along k) are fully coalesced.
// ---------------------------------------------------------------------------
__global__ __launch_bounds__(256)
void wt_kernel(const float* __restrict__ w, bf16_t* __restrict__ out) {
  __shared__ float tile[32][33];
  const int k0 = blockIdx.x * 32, n0 = blockIdx.y * 32;
  const int tx = threadIdx.x & 31, ty = threadIdx.x >> 5;   // 32 x 8
#pragma unroll
  for (int r = 0; r < 32; r += 8)
    tile[ty + r][tx] = w[(size_t)(k0 + ty + r) * DM + n0 + tx];
  __syncthreads();
#pragma unroll
  for (int r = 0; r < 32; r += 8)
    out[(size_t)(n0 + ty + r) * DM + k0 + tx] = (bf16_t)tile[tx][ty + r];
}

// ---------------------------------------------------------------------------
// GEMM: C[M x 1024] = A[M x 1024] * W + bias.
//   AK:   0 = A is f32 (convert on the fly), 1 = A is bf16.
//   OUTM: bit0 -> bf16 C (LDS-coalesced), bit1 -> f32 C (LDS-coalesced),
//         bit2 -> per-head transposed bf16 V  (vt[(b*16+h)*64+d][n], packed b128)
// Wave tile: 16 rows x 128 cols (8 WMMAs per A fragment). Block: 4 waves along M.
// ---------------------------------------------------------------------------
template <int AK, int OUTM>
__global__ __launch_bounds__(128)
void gemm_kernel(const void* __restrict__ a_ptr,
                 const bf16_t* __restrict__ wt, const float* __restrict__ bias,
                 bf16_t* __restrict__ c_bf16, float* __restrict__ c_f32,
                 bf16_t* __restrict__ vt) {
  constexpr int CPITCH = 136;   // 128 + 8 pad
  __shared__ float ctile[(OUTM & 3) ? 4 * 16 * CPITCH : 1];

  const int lane = threadIdx.x & 31;
  const int wave = threadIdx.x >> 5;
  const int half = lane >> 4;          // K-half selector (16-bit A/B layouts)
  const int lcol = lane & 15;
  const int m0 = blockIdx.x * 64 + wave * 16;
  const int n0 = blockIdx.y * 128;

  v8f acc[8] = {};
  const int arow = m0 + lcol;          // A fragment: row = lane%16
  const float*  af = (const float*)a_ptr  + (size_t)arow * DM;
  const bf16_t* ab = (const bf16_t*)a_ptr + (size_t)arow * DM;

  for (int kc = 0; kc < DM; kc += 32) {
    v16bf a;
    if constexpr (AK == 0) {
      f32x8 r0 = *(const f32x8*)(af + kc + 8 * half);
      f32x8 r1 = *(const f32x8*)(af + kc + 16 + 8 * half);
#pragma unroll
      for (int i = 0; i < 8; ++i) { a[i] = (bf16_t)r0[i]; a[i + 8] = (bf16_t)r1[i]; }
      if (kc + 64 < DM) __builtin_prefetch(af + kc + 64, 0, 3);
    } else {
      v8bf r0 = *(const v8bf*)(ab + kc + 8 * half);
      v8bf r1 = *(const v8bf*)(ab + kc + 16 + 8 * half);
      a = concat8(r0, r1);
      if (kc + 64 < DM) __builtin_prefetch(ab + kc + 64, 0, 3);
    }
#pragma unroll
    for (int t = 0; t < 8; ++t) {
      const int n = n0 + t * 16 + lcol;           // B fragment: col = lane%16
      v16bf b = concat8(*(const v8bf*)(wt + (size_t)n * DM + kc + 8 * half),
                        *(const v8bf*)(wt + (size_t)n * DM + kc + 16 + 8 * half));
      acc[t] = wmma_bf16(a, b, acc[t]);
    }
  }

  if constexpr ((OUTM & 4) != 0) {
    // Per-head V^T: fragment rows r=0..7 are consecutive n -> one 16B store.
    const int row0 = m0 + 8 * half;
    const int bb = row0 >> 10, nn = row0 & 1023;
#pragma unroll
    for (int t = 0; t < 8; ++t) {
      const int col = n0 + t * 16 + lcol;
      const float bv = bias[col];
      const int hh = col >> 6, dd = col & 63;
      v8bf pk;
#pragma unroll
      for (int r = 0; r < 8; ++r) pk[r] = (bf16_t)(acc[t][r] + bv);
      *(v8bf*)(vt + (((size_t)(bb * NH + hh) * DK + dd) << 10) + nn) = pk;
    }
  }
  if constexpr ((OUTM & 3) != 0) {
    // Stage C tile in LDS (column-per-lane -> row-contiguous), then coalesced stores.
    float* ct = &ctile[wave * 16 * CPITCH];
#pragma unroll
    for (int t = 0; t < 8; ++t) {
      const float bv = bias[n0 + t * 16 + lcol];
#pragma unroll
      for (int r = 0; r < 8; ++r)
        ct[(8 * half + r) * CPITCH + t * 16 + lcol] = acc[t][r] + bv;
    }
    __syncthreads();
    const int rrow = lane >> 1;                 // 2 lanes per row
    const int colb = (lane & 1) * 64;           // 64 cols each
    const float* cr = ct + rrow * CPITCH + colb;
    const size_t gbase = (size_t)(m0 + rrow) * DM + n0 + colb;
    if constexpr ((OUTM & 1) != 0) {
#pragma unroll
      for (int j = 0; j < 64; j += 16) {
        v16bf pk;
#pragma unroll
        for (int e = 0; e < 16; ++e) pk[e] = (bf16_t)cr[j + e];
        *(v16bf*)(c_bf16 + gbase + j) = pk;
      }
    }
    if constexpr ((OUTM & 2) != 0) {
#pragma unroll
      for (int j = 0; j < 64; j += 8) {
        f32x8 v;
#pragma unroll
        for (int e = 0; e < 8; ++e) v[e] = cr[j + e];
        *(f32x8*)(c_f32 + gbase + j) = v;
      }
    }
  }
}

// ---------------------------------------------------------------------------
// Flash attention + adaptive language logit, TRANSPOSED score form:
//   S^T = K Q^T  (A = K tile, B = Q^T)  ->  each lane owns ONE query column,
//   so softmax max/sum are in-lane reductions + a single shfl_xor(16).
//   O^T = V^T P^T (A = V^T from vt, B = P^T staged via 2 LDS stores/1 load).
// One wave per (b, h, 16-query tile).
// ---------------------------------------------------------------------------
__global__ __launch_bounds__(128)
void attn_kernel(const bf16_t* __restrict__ qp, const bf16_t* __restrict__ kp,
                 const bf16_t* __restrict__ vt, const float* __restrict__ sp,
                 bf16_t* __restrict__ ctx) {
  __shared__ bf16_t plds[4][16 * 32];      // per-wave P^T staging [q][key]
  const int lane = threadIdx.x & 31;
  const int wave = threadIdx.x >> 5;
  const int half = lane >> 4;
  const int lcol = lane & 15;              // this lane's query column
  const int bh = blockIdx.x;               // 0..63
  const int b  = bh >> 4, h = bh & 15;
  const int qt = blockIdx.y * 4 + wave;    // 0..63
  const int qb = b * 1024 + qt * 16;       // row base into [4096][1024]
  const int dcol = h * DK;
  bf16_t* pl = plds[wave];

  // Q^T B-fragments: lane holds column q = qb+lcol, contiguous d-run per chunk
  v16bf bq[2];
#pragma unroll
  for (int c = 0; c < 2; ++c)
    bq[c] = *(const v16bf*)(qp + (size_t)(qb + lcol) * DM + dcol + 32 * c + 16 * half);

  v8f o[4] = {};                           // O^T: 4 d-tiles x (8 rows/lane)
  float m = -INFINITY, s = 0.0f;           // per-query running stats (scalar!)

  const bf16_t* kbase = kp + (size_t)(b * 1024) * DM + dcol;
  for (int n0 = 0; n0 < 1024; n0 += 32) {
    // ---- S^T = K Q^T : two 16-key subtiles ----
    v8f sc[2];
#pragma unroll
    for (int u = 0; u < 2; ++u) {
      v8f acc = {};
#pragma unroll
      for (int c = 0; c < 2; ++c) {
        const bf16_t* kr = kbase + (size_t)(n0 + 16 * u + lcol) * DM + 32 * c;
        v16bf a = concat8(*(const v8bf*)(kr + 16 * half),
                          *(const v8bf*)(kr + 16 * half + 8));
        acc = wmma_bf16(a, bq[c], acc);
      }
      sc[u] = acc;   // sc[u][r] = score(key = n0+16u+8*half+r, q = lcol)
    }
    // ---- online softmax: in-lane reduce + one xor-16 shuffle ----
    float bm = fmaxf(sc[0][0], sc[1][0]);
#pragma unroll
    for (int r = 1; r < 8; ++r) bm = fmaxf(bm, fmaxf(sc[0][r], sc[1][r]));
    bm *= SCALE;
    bm = fmaxf(bm, __shfl_xor(bm, 16, 32));
    const float mn = fmaxf(m, bm);
    const float alpha = __expf(m - mn);
    float bs = 0.0f;
#pragma unroll
    for (int u = 0; u < 2; ++u) {
      v8bf pk;
#pragma unroll
      for (int r = 0; r < 8; ++r) {
        const float p = __expf(sc[u][r] * SCALE - mn);
        bs += p;
        pk[r] = (bf16_t)p;
      }
      *(v8bf*)(pl + lcol * 32 + 16 * u + 8 * half) = pk;   // P^T[q][keys]
    }
    bs += __shfl_xor(bs, 16, 32);
    s = s * alpha + bs;
    m = mn;
#pragma unroll
    for (int t = 0; t < 4; ++t)
#pragma unroll
      for (int r = 0; r < 8; ++r) o[t][r] *= alpha;

    // ---- O^T += V^T P^T  (same-wave DS ops are in-order) ----
    v16bf bp = *(const v16bf*)(pl + lcol * 32 + 16 * half);  // single B-frag
#pragma unroll
    for (int t = 0; t < 4; ++t) {
      const bf16_t* vr = vt + (((size_t)bh * DK + 16 * t + lcol) << 10) + n0;
      v16bf a = concat8(*(const v8bf*)(vr + 16 * half),
                        *(const v8bf*)(vr + 16 * half + 8));
      o[t] = wmma_bf16(a, bp, o[t]);
    }
  }

  // ---- adaptive language logit (reuses in-register Q^T fragment) ----
  const float* srow = sp + (size_t)(qb + lcol) * DM + dcol;
  float part = 0.0f;
#pragma unroll
  for (int c = 0; c < 2; ++c) {
    f32x8 s0 = *(const f32x8*)(srow + 32 * c + 16 * half);
    f32x8 s1 = *(const f32x8*)(srow + 32 * c + 16 * half + 8);
#pragma unroll
    for (int e = 0; e < 8; ++e)
      part += (float)bq[c][e] * s0[e] + (float)bq[c][e + 8] * s1[e];
  }
  part += __shfl_xor(part, 16, 32);
  const float lng = part * SCALE;
  const float mn = fmaxf(m, lng);
  const float alphaf = __expf(m - mn);
  const float pL = __expf(lng - mn);
  const float inv = 1.0f / (s * alphaf + pL);

  // ---- finalize + packed 16B stores (8 consecutive d per lane) ----
  bf16_t* crow = ctx + (size_t)(qb + lcol) * DM + dcol;
#pragma unroll
  for (int t = 0; t < 4; ++t) {
    const int d0 = 16 * t + 8 * half;       // o[t] row r -> d = d0 + r
    f32x8 sv = *(const f32x8*)(srow + d0);
    v8bf outv;
#pragma unroll
    for (int r = 0; r < 8; ++r)
      outv[r] = (bf16_t)((o[t][r] * alphaf + pL * sv[r]) * inv);
    *(v8bf*)(crow + d0) = outv;
  }
}

// ---------------------------------------------------------------------------
extern "C" void kernel_launch(void* const* d_in, const int* in_sizes, int n_in,
                              void* d_out, int out_size, void* d_ws, size_t ws_size,
                              hipStream_t stream) {
  const float* queries = (const float*)d_in[0];
  const float* keys    = (const float*)d_in[1];
  const float* values  = (const float*)d_in[2];
  const float* lang    = (const float*)d_in[3];
  const float* Wq = (const float*)d_in[4];  const float* b_q = (const float*)d_in[5];
  const float* Wk = (const float*)d_in[6];  const float* b_k = (const float*)d_in[7];
  const float* Wv = (const float*)d_in[8];  const float* b_v = (const float*)d_in[9];
  const float* Ws = (const float*)d_in[10]; const float* b_s = (const float*)d_in[11];
  const float* Wo = (const float*)d_in[12]; const float* b_o = (const float*)d_in[13];

  const size_t MAT = (size_t)DM * DM;              // 1M elements
  char* ws = (char*)d_ws;
  bf16_t* wt  = (bf16_t*)ws;                       // 5 x 1M bf16  (10 MB)
  bf16_t* qp  = (bf16_t*)(ws + 5 * MAT * 2);       // 4096x1024 bf16 (8 MB)
  bf16_t* kp  = qp + (size_t)NROW * DM;            // 8 MB
  float*  spf = (float*)(kp + (size_t)NROW * DM);  // 16 MB (f32 language proj)
  bf16_t* vt  = (bf16_t*)(spf + (size_t)NROW * DM);// 8 MB (per-head V^T)
  bf16_t* ctx = vt + (size_t)NROW * DM;            // 8 MB

  // 1) weights -> transposed bf16 (LDS-tiled, coalesced both ways)
  dim3 tg(DM / 32, DM / 32);
  wt_kernel<<<tg, 256, 0, stream>>>(Wq, wt + 0 * MAT);
  wt_kernel<<<tg, 256, 0, stream>>>(Wk, wt + 1 * MAT);
  wt_kernel<<<tg, 256, 0, stream>>>(Wv, wt + 2 * MAT);
  wt_kernel<<<tg, 256, 0, stream>>>(Ws, wt + 3 * MAT);
  wt_kernel<<<tg, 256, 0, stream>>>(Wo, wt + 4 * MAT);

  // 2) projections (WMMA GEMMs), 16x128 wave tiles
  dim3 gg(NROW / 64, DM / 128), gb(128);
  gemm_kernel<0, 1><<<gg, gb, 0, stream>>>(queries, wt + 0 * MAT, b_q, qp, nullptr, nullptr);
  gemm_kernel<0, 1><<<gg, gb, 0, stream>>>(keys,    wt + 1 * MAT, b_k, kp, nullptr, nullptr);
  gemm_kernel<0, 4><<<gg, gb, 0, stream>>>(values,  wt + 2 * MAT, b_v, nullptr, nullptr, vt);
  gemm_kernel<0, 2><<<gg, gb, 0, stream>>>(lang,    wt + 3 * MAT, b_s, nullptr, spf, nullptr);

  // 3) flash attention with adaptive language column (transposed-score form)
  attn_kernel<<<dim3(64, 16), 128, 0, stream>>>(qp, kp, vt, spf, ctx);

  // 4) output projection -> f32 d_out
  gemm_kernel<1, 2><<<gg, gb, 0, stream>>>(ctx, wt + 4 * MAT, b_o,
                                           nullptr, (float*)d_out, nullptr);
}
